// Precise45KCNN_QNN_Model_22033182228573
// MI455X (gfx1250) — matrix operand order repaired and checked by reading, not compiled, not run on hardware
//
#include <hip/hip_runtime.h>
#include <math.h>

typedef __attribute__((ext_vector_type(16))) _Float16 v16h;
typedef __attribute__((ext_vector_type(8)))  float    v8f;

#define TILE     16
#define NTHREADS 256

// ---- LDS layout (float offsets), phases reuse regions ----
#define OFF_C1    0        // conv1 out: 8*14*14 = 1568
#define OFF_C2    1568     // conv2 out: 16*7*7 =  784
#define OFF_C3    2352     // conv3 out: 32*7*7 = 1568   (scratch ends 3920)
#define OFF_FLAT  4096     // pooled  : 16*288  = 4608
#define OFF_XIN   8704     // staged input image: 784 (reuses FEAT region in conv phase)
#define OFF_FEAT  8704     // fc out  : 16*256  = 4096
#define OFF_STATE 0        // qstate  : 16*256  = 4096  (reuses conv scratch)
#define OFF_QV    12800    // 16 scalars (invnorm, then <Z0>)
#define OFF_H1    0        // 16*512 = 8192 (reuses state+flat, both dead)
#define OFF_H2    8704     // 16*256 = 4096 (reuses feat)
#define OFF_H3    0        // 16*128 = 2048
#define OFF_H4    2048     // 16*64  = 1024
#define SMEM_FLOATS 12816  // 51264 bytes

// One wave computes a 16x16 output tile of  Out = relu(A[16xK] * W[N,K]^T + b)
// using v_wmma_f32_16x16x32_f16 with fully unrolled K (compile-time).
// A fragment per lane = two contiguous 32B runs from LDS (ds_load_b128 x4).
// B fragment per lane = two contiguous 32B runs of the weight row (global_load_b128 x4).
// Layouts per CDNA5 ISA 7.12.2:
//   A 16x32 f16 : lanes0-15 K={kk..kk+7, kk+16..kk+23}; lanes16-31 K={+8, +24}
//   B 32x16 f16 : lanes0-15 K={kk..kk+7, kk+16..kk+23}; lanes16-31 K={+8, +24} (col = lane%16)
//   D f32       : element i -> M = i + 8*(lane/16), N = lane%16
template <int K>
__device__ __forceinline__ void gemm_tile_wmma(
    const float* __restrict__ Alds, int lda,
    const float* __restrict__ W, const float* __restrict__ bias,
    int n0, float* __restrict__ Olds, int ldo, int lane)
{
  const int grp = lane >> 4;
  const int r   = lane & 15;
  const float4* __restrict__ Arow = (const float4*)(Alds + r * lda);
  const float4* __restrict__ Wrow = (const float4*)(W + (size_t)(n0 + r) * K);
  v8f acc = {};
#pragma unroll
  for (int kk = 0; kk < K; kk += 32) {
    const int ab = (kk + grp * 8) >> 2;     // float4 index of A run start
    const float4 a0 = Arow[ab];
    const float4 a1 = Arow[ab + 1];
    const float4 a2 = Arow[ab + 4];         // second run at +16 floats
    const float4 a3 = Arow[ab + 5];
    const int bb = (kk + grp * 8) >> 2;     // float4 index of B run start
    const float4 b0 = Wrow[bb];
    const float4 b1 = Wrow[bb + 1];
    const float4 b2 = Wrow[bb + 4];
    const float4 b3 = Wrow[bb + 5];
    v16h a, b;
    a[0]  = (_Float16)a0.x; a[1]  = (_Float16)a0.y; a[2]  = (_Float16)a0.z; a[3]  = (_Float16)a0.w;
    a[4]  = (_Float16)a1.x; a[5]  = (_Float16)a1.y; a[6]  = (_Float16)a1.z; a[7]  = (_Float16)a1.w;
    a[8]  = (_Float16)a2.x; a[9]  = (_Float16)a2.y; a[10] = (_Float16)a2.z; a[11] = (_Float16)a2.w;
    a[12] = (_Float16)a3.x; a[13] = (_Float16)a3.y; a[14] = (_Float16)a3.z; a[15] = (_Float16)a3.w;
    b[0]  = (_Float16)b0.x; b[1]  = (_Float16)b0.y; b[2]  = (_Float16)b0.z; b[3]  = (_Float16)b0.w;
    b[4]  = (_Float16)b1.x; b[5]  = (_Float16)b1.y; b[6]  = (_Float16)b1.z; b[7]  = (_Float16)b1.w;
    b[8]  = (_Float16)b2.x; b[9]  = (_Float16)b2.y; b[10] = (_Float16)b2.z; b[11] = (_Float16)b2.w;
    b[12] = (_Float16)b3.x; b[13] = (_Float16)b3.y; b[14] = (_Float16)b3.z; b[15] = (_Float16)b3.w;
    acc = __builtin_amdgcn_wmma_f32_16x16x32_f16(
        false, a, false, b, (short)0, acc, false, false);
  }
  const float bv = bias[n0 + r];
#pragma unroll
  for (int i = 0; i < 8; ++i) {
    int m = i + 8 * grp;
    Olds[m * ldo + n0 + r] = fmaxf(acc[i] + bv, 0.0f);
  }
}

__global__ __launch_bounds__(NTHREADS)
void fused_cnn_qnn(const float* __restrict__ x,
                   const float* __restrict__ w1, const float* __restrict__ b1,
                   const float* __restrict__ w2, const float* __restrict__ b2,
                   const float* __restrict__ w3, const float* __restrict__ b3,
                   const float* __restrict__ fcw, const float* __restrict__ fcb,
                   const float* __restrict__ qw,
                   const float* __restrict__ c1w, const float* __restrict__ c1b,
                   const float* __restrict__ c2w, const float* __restrict__ c2b,
                   const float* __restrict__ c3w, const float* __restrict__ c3b,
                   const float* __restrict__ c4w, const float* __restrict__ c4b,
                   const float* __restrict__ c5w, const float* __restrict__ c5b,
                   float* __restrict__ out, int Btot)
{
  __shared__ __align__(16) float sm[SMEM_FLOATS];
  const int tid  = threadIdx.x;
  const int lane = tid & 31;
  const int wave = tid >> 5;

  if (tid == 0) {  // warm L2 with the dense-layer weights (global_prefetch_b8)
    __builtin_prefetch(fcw, 0, 1);
    __builtin_prefetch(c2w, 0, 1);
    __builtin_prefetch(c3w, 0, 1);
    __builtin_prefetch(c4w, 0, 1);
  }

  // ================= conv pipeline, one image at a time in LDS ==============
  for (int img = 0; img < TILE; ++img) {
    long gi = (long)blockIdx.x * TILE + img;
    if (gi >= Btot) gi = Btot - 1;                 // safe clamp (tail)
    const float4* __restrict__ xin4 = (const float4*)(x + (size_t)gi * 784);

    // stage image in LDS (784 floats = 196 float4, one coalesced pass)
    for (int o = tid; o < 196; o += NTHREADS)
      ((float4*)(sm + OFF_XIN))[o] = xin4[o];
    __syncthreads();

    // conv1: 1->8ch, 5x5, s2, p2, 28->14, relu
    for (int o = tid; o < 8 * 196; o += NTHREADS) {
      int c = o / 196, rem = o - c * 196, oy = rem / 14, ox = rem - oy * 14;
      float acc = b1[c];
      for (int ky = 0; ky < 5; ++ky) {
        int iy = 2 * oy - 2 + ky;
        if (iy < 0 || iy >= 28) continue;
        for (int kx = 0; kx < 5; ++kx) {
          int ix = 2 * ox - 2 + kx;
          if (ix < 0 || ix >= 28) continue;
          acc += sm[OFF_XIN + iy * 28 + ix] * w1[c * 25 + ky * 5 + kx];
        }
      }
      sm[OFF_C1 + o] = fmaxf(acc, 0.0f);
    }
    __syncthreads();

    // conv2: 8->16ch, 3x3, s2, p1, 14->7, relu
    for (int o = tid; o < 16 * 49; o += NTHREADS) {
      int c = o / 49, rem = o - c * 49, oy = rem / 7, ox = rem - oy * 7;
      float acc = b2[c];
      for (int ic = 0; ic < 8; ++ic)
        for (int ky = 0; ky < 3; ++ky) {
          int iy = 2 * oy - 1 + ky;
          if (iy < 0 || iy >= 14) continue;
          for (int kx = 0; kx < 3; ++kx) {
            int ix = 2 * ox - 1 + kx;
            if (ix < 0 || ix >= 14) continue;
            acc += sm[OFF_C1 + ic * 196 + iy * 14 + ix] *
                   w2[(c * 8 + ic) * 9 + ky * 3 + kx];
          }
        }
      sm[OFF_C2 + o] = fmaxf(acc, 0.0f);
    }
    __syncthreads();

    // conv3: 16->32ch, 3x3, s1, p1, 7x7, relu
    for (int o = tid; o < 32 * 49; o += NTHREADS) {
      int c = o / 49, rem = o - c * 49, oy = rem / 7, ox = rem - oy * 7;
      float acc = b3[c];
      for (int ic = 0; ic < 16; ++ic)
        for (int ky = 0; ky < 3; ++ky) {
          int iy = oy - 1 + ky;
          if (iy < 0 || iy >= 7) continue;
          for (int kx = 0; kx < 3; ++kx) {
            int ix = ox - 1 + kx;
            if (ix < 0 || ix >= 7) continue;
            acc += sm[OFF_C2 + ic * 49 + iy * 7 + ix] *
                   w3[(c * 16 + ic) * 9 + ky * 3 + kx];
          }
        }
      sm[OFF_C3 + o] = fmaxf(acc, 0.0f);
    }
    __syncthreads();

    // adaptive avg pool 7->3 (bins start at 0,2,4, each 3 wide), flatten
    for (int o = tid; o < 288; o += NTHREADS) {
      int c = o / 9, rem = o - c * 9, oy = rem / 3, ox = rem - oy * 3;
      int hs = 2 * oy, ws = 2 * ox;
      float s = 0.0f;
      for (int dy = 0; dy < 3; ++dy)
        for (int dx = 0; dx < 3; ++dx)
          s += sm[OFF_C3 + c * 49 + (hs + dy) * 7 + (ws + dx)];
      sm[OFF_FLAT + img * 288 + o] = s * (1.0f / 9.0f);
    }
    __syncthreads();
  }

  // ================= fc 288->256 via WMMA (16 N-tiles, 2 per wave) ==========
  for (int t = wave; t < 16; t += 8)
    gemm_tile_wmma<288>(&sm[OFF_FLAT], 288, fcw, fcb, t * 16,
                        &sm[OFF_FEAT], 256, lane);
  __syncthreads();

  // ================= L2-normalize -> quantum statevector ====================
  if (tid < TILE) {
    const float* f = &sm[OFF_FEAT + tid * 256];
    float ss = 0.0f;
    for (int i = 0; i < 256; ++i) ss += f[i] * f[i];
    sm[OFF_QV + tid] = 1.0f / fmaxf(sqrtf(ss), 1e-12f);
  }
  __syncthreads();
  for (int t = tid; t < TILE * 256; t += NTHREADS) {
    int img = t >> 8;
    sm[OFF_STATE + t] = sm[OFF_FEAT + t] * sm[OFF_QV + img];
  }
  __syncthreads();

  // ================= 7 layers x (8 RY + 7 CNOT), wire0 = MSB ================
  for (int l = 0; l < 7; ++l) {
    for (int q = 0; q < 8; ++q) {
      float th = 0.5f * qw[l * 8 + q];
      float ct = cosf(th), sn = sinf(th);
      int b = 7 - q, mask = 1 << b;
      for (int t = tid; t < TILE * 128; t += NTHREADS) {
        int img = t >> 7, p = t & 127;
        int i0 = ((p >> b) << (b + 1)) | (p & (mask - 1));
        int i1 = i0 | mask;
        float a0 = sm[OFF_STATE + img * 256 + i0];
        float a1 = sm[OFF_STATE + img * 256 + i1];
        sm[OFF_STATE + img * 256 + i0] = ct * a0 - sn * a1;
        sm[OFF_STATE + img * 256 + i1] = sn * a0 + ct * a1;
      }
      __syncthreads();
    }
    for (int q = 0; q < 7; ++q) {          // CNOT(q, q+1): bc = bt+1
      int bt = 6 - q;
      for (int t = tid; t < TILE * 64; t += NTHREADS) {
        int img = t >> 6, p = t & 63;
        int i0 = ((p >> bt) << (bt + 2)) | (2 << bt) | (p & ((1 << bt) - 1));
        int i1 = i0 | (1 << bt);
        float a0 = sm[OFF_STATE + img * 256 + i0];
        float a1 = sm[OFF_STATE + img * 256 + i1];
        sm[OFF_STATE + img * 256 + i0] = a1;
        sm[OFF_STATE + img * 256 + i1] = a0;
      }
      __syncthreads();
    }
  }

  // <Z0> = sum(MSB=0 amps^2) - sum(MSB=1 amps^2)
  if (tid < TILE) {
    const float* st = &sm[OFF_STATE + tid * 256];
    float acc = 0.0f;
    for (int i = 0; i < 128; ++i)   acc += st[i] * st[i];
    for (int i = 128; i < 256; ++i) acc -= st[i] * st[i];
    sm[OFF_QV + tid] = acc;
  }
  __syncthreads();

  // ================= classical head =========================================
  // c1: 1 -> 512 (outer product, VALU)
  for (int t = tid; t < TILE * 512; t += NTHREADS) {
    int img = t >> 9, n = t & 511;
    sm[OFF_H1 + t] = fmaxf(sm[OFF_QV + img] * c1w[n] + c1b[n], 0.0f);
  }
  __syncthreads();
  // c2: 512 -> 256 (WMMA, 16 tiles)
  for (int t = wave; t < 16; t += 8)
    gemm_tile_wmma<512>(&sm[OFF_H1], 512, c2w, c2b, t * 16, &sm[OFF_H2], 256, lane);
  __syncthreads();
  // c3: 256 -> 128 (WMMA, 8 tiles)
  if (wave < 8)
    gemm_tile_wmma<256>(&sm[OFF_H2], 256, c3w, c3b, wave * 16, &sm[OFF_H3], 128, lane);
  __syncthreads();
  // c4: 128 -> 64 (WMMA, 4 tiles)
  if (wave < 4)
    gemm_tile_wmma<128>(&sm[OFF_H3], 128, c4w, c4b, wave * 16, &sm[OFF_H4], 64, lane);
  __syncthreads();
  // c5: 64 -> 1 + sigmoid
  if (tid < TILE) {
    long gi = (long)blockIdx.x * TILE + tid;
    if (gi < Btot) {
      const float* h = &sm[OFF_H4 + tid * 64];
      float z = c5b[0];
      for (int i = 0; i < 64; ++i) z += h[i] * c5w[i];
      out[gi] = 1.0f / (1.0f + expf(-z));
    }
  }
}

extern "C" void kernel_launch(void* const* d_in, const int* in_sizes, int n_in,
                              void* d_out, int out_size, void* d_ws, size_t ws_size,
                              hipStream_t stream) {
  const float* x   = (const float*)d_in[0];
  const float* w1  = (const float*)d_in[1];
  const float* b1  = (const float*)d_in[2];
  const float* w2  = (const float*)d_in[3];
  const float* b2  = (const float*)d_in[4];
  const float* w3  = (const float*)d_in[5];
  const float* b3  = (const float*)d_in[6];
  const float* fcw = (const float*)d_in[7];
  const float* fcb = (const float*)d_in[8];
  const float* qw  = (const float*)d_in[9];
  const float* c1w = (const float*)d_in[10];
  const float* c1b = (const float*)d_in[11];
  const float* c2w = (const float*)d_in[12];
  const float* c2b = (const float*)d_in[13];
  const float* c3w = (const float*)d_in[14];
  const float* c3b = (const float*)d_in[15];
  const float* c4w = (const float*)d_in[16];
  const float* c4b = (const float*)d_in[17];
  const float* c5w = (const float*)d_in[18];
  const float* c5b = (const float*)d_in[19];

  int B = in_sizes[0] / 784;                  // 8192
  int blocks = (B + TILE - 1) / TILE;         // 512
  fused_cnn_qnn<<<blocks, NTHREADS, 0, stream>>>(
      x, w1, b1, w2, b2, w3, b3, fcw, fcb, qw,
      c1w, c1b, c2w, c2b, c3w, c3b, c4w, c4b, c5w, c5b,
      (float*)d_out, B);
}
